// GraphConv_82695300317343
// MI455X (gfx1250) — matrix operand order repaired and checked by reading, not compiled, not run on hardware
//
#include <hip/hip_runtime.h>

typedef __attribute__((ext_vector_type(16))) _Float16 v16h;
typedef __attribute__((ext_vector_type(8)))  float    v8f;

#define DEVFN __device__ __forceinline__

// ---------- order-preserving float<->uint encoding for atomic segment-max ----
DEVFN unsigned enc_f32(float f) {
  unsigned u = __float_as_uint(f);
  return (u & 0x80000000u) ? ~u : (u | 0x80000000u);
}
DEVFN float dec_f32(unsigned e) {
  unsigned u = (e & 0x80000000u) ? (e & 0x7fffffffu) : ~e;
  return __uint_as_float(u);
}

// ---------------------------- utility kernels -------------------------------
__global__ void k_zero(float* __restrict__ p, int n) {
  int i = blockIdx.x * blockDim.x + threadIdx.x;
  const int st = gridDim.x * blockDim.x;
  for (; i < n; i += st) p[i] = 0.0f;
}

// mean of edge_attr over E rows (3 columns), accumulated pre-scaled by 1/E
__global__ void k_ea_mean(const float* __restrict__ ea, int E, float invE,
                          float* __restrict__ mean3) {
  __shared__ float sh0[256], sh1[256], sh2[256];
  float s0 = 0.f, s1 = 0.f, s2 = 0.f;
  for (int i = blockIdx.x * blockDim.x + threadIdx.x; i < E;
       i += gridDim.x * blockDim.x) {
    s0 += ea[3 * i + 0];
    s1 += ea[3 * i + 1];
    s2 += ea[3 * i + 2];
  }
  const int t = threadIdx.x;
  sh0[t] = s0; sh1[t] = s1; sh2[t] = s2;
  __syncthreads();
  for (int w = 128; w > 0; w >>= 1) {
    if (t < w) { sh0[t] += sh0[t + w]; sh1[t] += sh1[t + w]; sh2[t] += sh2[t + w]; }
    __syncthreads();
  }
  if (t == 0) {
    atomicAdd(&mean3[0], sh0[0] * invE);
    atomicAdd(&mean3[1], sh1[0] * invE);
    atomicAdd(&mean3[2], sh2[0] * invE);
  }
}

// ------------------- dual GEMM (X@Wl, X@Wr) via WMMA f16 --------------------
// X: [nrows, K] f32 row-major.  Wl/Wr: [K, OUT] f32 row-major.
// Each wave owns one 16-row tile; B fragments loaded once, reused grid-stride.
// A layout (16-bit A 16x32): lane L -> M = L&15, half h = L>>4; packed elem j:
//   j in [0,8):  K = 32*chunk + 8*h + j          (VGPRs 0..3)
//   j in [8,16): K = 32*chunk + 16 + 8*h + (j-8) (VGPRs 4..7)
// B layout (16-bit B 32x16): lane L -> N = L&15, half h = L>>4; elem j:
//   same K mapping as A (8*h + j, then 16 + 8*h + (j-8))
// C/D layout (f32 16x16): lane L -> N = L&15; VGPR r -> M = 8*(L>>4) + r
template <int K, int OUT>
__launch_bounds__(256)
__global__ void k_gemm_dual_wmma(const float* __restrict__ X,
                                 const float* __restrict__ Wl,
                                 const float* __restrict__ Wr,
                                 float* __restrict__ XL,
                                 float* __restrict__ XR, int nrows) {
  constexpr int NC = (K + 31) / 32;    // K chunks of 32
  constexpr int NT = (OUT + 15) / 16;  // N tiles of 16
  const int lane = threadIdx.x & 31;
  const int wavesPerBlock = blockDim.x >> 5;
  const int waveGlobal = blockIdx.x * wavesPerBlock + (threadIdx.x >> 5);
  const int totalWaves = gridDim.x * wavesPerBlock;
  const int n  = lane & 15;
  const int hh = lane >> 4;

  // B fragments (weights): tiny, L1-resident; load once per wave.
  v16h bl[NC][NT], br[NC][NT];
  for (int c = 0; c < NC; ++c) {
    for (int t = 0; t < NT; ++t) {
      const int col = t * 16 + n;
      v16h fl = {}, fr = {};
#pragma unroll
      for (int j = 0; j < 16; ++j) {
        const int kk = c * 32 + hh * 8 + j + (j >= 8 ? 8 : 0);
        float vl = 0.f, vr = 0.f;
        if (col < OUT && kk < K) {
          vl = Wl[kk * OUT + col];
          vr = Wr[kk * OUT + col];
        }
        fl[j] = (_Float16)vl;
        fr[j] = (_Float16)vr;
      }
      bl[c][t] = fl;
      br[c][t] = fr;
    }
  }

  const int ntiles = (nrows + 15) >> 4;
  for (int tile = waveGlobal; tile < ntiles; tile += totalWaves) {
    const int row = tile * 16 + n;  // A-matrix M index
    v8f accl[NT], accr[NT];
    for (int t = 0; t < NT; ++t) { v8f z = {}; accl[t] = z; accr[t] = z; }

    for (int c = 0; c < NC; ++c) {
      v16h a = {};
      const int kb = c * 32 + hh * 8;
      if (row < nrows) {
        const float* xp = X + (size_t)row * K;
#pragma unroll
        for (int j = 0; j < 8; ++j) {
          const int k0 = kb + j;        // contiguous run 1 (VGPRs 0..3)
          const int k1 = kb + 16 + j;   // contiguous run 2 (VGPRs 4..7)
          a[j]     = (k0 < K) ? (_Float16)xp[k0] : (_Float16)0.f;
          a[j + 8] = (k1 < K) ? (_Float16)xp[k1] : (_Float16)0.f;
        }
      }
#pragma unroll
      for (int t = 0; t < NT; ++t) {
        accl[t] = __builtin_amdgcn_wmma_f32_16x16x32_f16(
            false, a, false, bl[c][t], (short)0, accl[t], false, false);
        accr[t] = __builtin_amdgcn_wmma_f32_16x16x32_f16(
            false, a, false, br[c][t], (short)0, accr[t], false, false);
      }
    }

    for (int t = 0; t < NT; ++t) {
      const int col = t * 16 + n;
      if (col < OUT) {
#pragma unroll
        for (int r = 0; r < 8; ++r) {
          const int rr = tile * 16 + hh * 8 + r;
          if (rr < nrows) {
            XL[(size_t)rr * OUT + col] = accl[t][r];
            XR[(size_t)rr * OUT + col] = accr[t][r];
          }
        }
      }
    }
  }
}

// ---------------- edge pass A: scores + segment max (atomic) ----------------
template <int H, int C>
__launch_bounds__(256)
__global__ void k_edge_score(const int* __restrict__ src, const int* __restrict__ dst,
                             const float* __restrict__ eattr,
                             const float* __restrict__ mean3,
                             const float* __restrict__ XL, const float* __restrict__ XR,
                             const float* __restrict__ We, const float* __restrict__ att,
                             float* __restrict__ sbuf, unsigned* __restrict__ smax,
                             int E, int Etot) {
  constexpr int HC = H * C;
  __shared__ float sWe[3 * HC];
  __shared__ float sAtt[HC];
  for (int i = threadIdx.x; i < 3 * HC; i += blockDim.x) sWe[i] = We[i];
  for (int i = threadIdx.x; i < HC; i += blockDim.x) sAtt[i] = att[i];
  __syncthreads();

  for (int e = blockIdx.x * blockDim.x + threadIdx.x; e < Etot;
       e += gridDim.x * blockDim.x) {
    int s, d;
    float e0, e1, e2;
    if (e < E) {
      s = src[e]; d = dst[e];
      e0 = eattr[3 * (size_t)e]; e1 = eattr[3 * (size_t)e + 1]; e2 = eattr[3 * (size_t)e + 2];
    } else {
      s = d = e - E;
      e0 = mean3[0]; e1 = mean3[1]; e2 = mean3[2];
    }
    float sc[H];
#pragma unroll
    for (int h = 0; h < H; ++h) sc[h] = 0.f;
    const float* xlp = XL + (size_t)s * HC;
    const float* xrp = XR + (size_t)d * HC;
#pragma unroll
    for (int c = 0; c < HC; ++c) {
      float m = xlp[c] + xrp[c] + e0 * sWe[c] + e1 * sWe[HC + c] + e2 * sWe[2 * HC + c];
      m = (m > 0.f) ? m : 0.2f * m;  // LeakyReLU(0.2)
      sc[c / C] += sAtt[c] * m;
    }
#pragma unroll
    for (int h = 0; h < H; ++h) {
      sbuf[(size_t)e * H + h] = sc[h];
      atomicMax(&smax[d * H + h], enc_f32(sc[h]));
    }
  }
}

// ---------------- edge pass B: exp(s - max) + segment sum -------------------
template <int H>
__launch_bounds__(256)
__global__ void k_edge_exp(const int* __restrict__ dst,
                           const unsigned* __restrict__ smax,
                           float* __restrict__ sbuf, float* __restrict__ den,
                           int E, int Etot) {
  for (int e = blockIdx.x * blockDim.x + threadIdx.x; e < Etot;
       e += gridDim.x * blockDim.x) {
    const int d = (e < E) ? dst[e] : (e - E);
#pragma unroll
    for (int h = 0; h < H; ++h) {
      const float sm = dec_f32(smax[d * H + h]);
      const float ex = __expf(sbuf[(size_t)e * H + h] - sm);
      sbuf[(size_t)e * H + h] = ex;
      atomicAdd(&den[d * H + h], ex);
    }
  }
}

// ---------------- edge pass C: scatter-add xl[src] * alpha ------------------
template <int H, int C>
__launch_bounds__(256)
__global__ void k_edge_aggr(const int* __restrict__ src, const int* __restrict__ dst,
                            const float* __restrict__ XL,
                            const float* __restrict__ sbuf,
                            const float* __restrict__ den,
                            float* __restrict__ acc, int E, int Etot) {
  constexpr int HC = H * C;
  for (int e = blockIdx.x * blockDim.x + threadIdx.x; e < Etot;
       e += gridDim.x * blockDim.x) {
    int s, d;
    if (e < E) { s = src[e]; d = dst[e]; } else { s = d = e - E; }
    float alpha[H];
#pragma unroll
    for (int h = 0; h < H; ++h)
      alpha[h] = sbuf[(size_t)e * H + h] / den[d * H + h];
    const float* xlp = XL + (size_t)s * HC;
    float* ap = acc + (size_t)d * HC;
#pragma unroll
    for (int c = 0; c < HC; ++c) atomicAdd(&ap[c], xlp[c] * alpha[c / C]);
  }
}

// ---------------- bias + relu (layer 1 output) ------------------------------
__global__ void k_bias_relu(const float* __restrict__ acc, const float* __restrict__ b,
                            float* __restrict__ out, int n, int ch) {
  const int total = n * ch;
  for (int i = blockIdx.x * blockDim.x + threadIdx.x; i < total;
       i += gridDim.x * blockDim.x) {
    const int c = i % ch;
    out[i] = fmaxf(acc[i] + b[c], 0.f);
  }
}

// ------- layer-2 bias + relu fused with global max pool (values >= 0) -------
__global__ void k_bias_relu_pool(const float* __restrict__ acc,
                                 const float* __restrict__ b,
                                 const int* __restrict__ batch,
                                 unsigned* __restrict__ pooled, int n) {
  const int total = n * 20;
  for (int i = blockIdx.x * blockDim.x + threadIdx.x; i < total;
       i += gridDim.x * blockDim.x) {
    const int node = i / 20, c = i % 20;
    const float v = fmaxf(acc[i] + b[c], 0.f);
    // post-ReLU values are non-negative: raw IEEE bits are monotone under u32
    atomicMax(&pooled[batch[node] * 20 + c], __float_as_uint(v));
  }
}

// ---------------- final MLP 20 -> 10 -> 1 over 128 graphs -------------------
__global__ void k_mlp(const unsigned* __restrict__ pooled,
                      const float* __restrict__ fw1, const float* __restrict__ fb1,
                      const float* __restrict__ fw2, const float* __restrict__ fb2,
                      float* __restrict__ out, int G) {
  const int g = blockIdx.x * blockDim.x + threadIdx.x;
  if (g >= G) return;
  float p[20];
#pragma unroll
  for (int k = 0; k < 20; ++k) p[k] = __uint_as_float(pooled[g * 20 + k]);
  float o = fb2[0];
#pragma unroll
  for (int j = 0; j < 10; ++j) {
    float hs = fb1[j];
#pragma unroll
    for (int k = 0; k < 20; ++k) hs += p[k] * fw1[k * 10 + j];
    o += fmaxf(hs, 0.f) * fw2[j];
  }
  out[g] = o;
}

// ---------------------------------------------------------------------------
static inline void zero_f(float* p, long long n, hipStream_t s) {
  int blocks = (int)((n + 255) / 256);
  if (blocks > 2048) blocks = 2048;
  if (blocks < 1) blocks = 1;
  k_zero<<<blocks, 256, 0, s>>>(p, (int)n);
}

extern "C" void kernel_launch(void* const* d_in, const int* in_sizes, int n_in,
                              void* d_out, int out_size, void* d_ws, size_t ws_size,
                              hipStream_t stream) {
  // inputs in setup_inputs order
  const float* x     = (const float*)d_in[0];   // [N, 64]
  const int*   ei    = (const int*)d_in[1];     // [2, E]
  const float* ea    = (const float*)d_in[2];   // [E, 3]
  const int*   batch = (const int*)d_in[3];     // [N]
  const float* Wl1 = (const float*)d_in[4];
  const float* Wr1 = (const float*)d_in[5];
  const float* We1 = (const float*)d_in[6];
  const float* att1 = (const float*)d_in[7];
  const float* b1  = (const float*)d_in[8];
  const float* Wl2 = (const float*)d_in[9];
  const float* Wr2 = (const float*)d_in[10];
  const float* We2 = (const float*)d_in[11];
  const float* att2 = (const float*)d_in[12];
  const float* b2  = (const float*)d_in[13];
  const float* fw1 = (const float*)d_in[14];
  const float* fb1 = (const float*)d_in[15];
  const float* fw2 = (const float*)d_in[16];
  const float* fb2 = (const float*)d_in[17];

  const int N = in_sizes[0] / 64;
  const int E = in_sizes[1] / 2;
  const int Etot = E + N;  // with self-loops
  const int* src = ei;
  const int* dst = ei + E;

  // workspace layout (floats)
  float* W = (float*)d_ws;
  size_t off = 0;
  float*    mean3  = W + off; off += 4;
  unsigned* smax   = (unsigned*)(W + off); off += (size_t)N * 4;
  float*    den    = W + off; off += (size_t)N * 4;
  float*    xl     = W + off; off += (size_t)N * 20;
  float*    xr     = W + off; off += (size_t)N * 20;
  float*    acc    = W + off; off += (size_t)N * 20;
  float*    h1     = W + off; off += (size_t)N * 9;
  float*    sbuf   = W + off; off += (size_t)Etot * 4;
  unsigned* pooled = (unsigned*)(W + off); off += 128 * 20;
  (void)ws_size; (void)n_in;

  const int EB = 2048;  // edge-pass grid (grid-stride)

  // ---- init ----
  zero_f(mean3, 4, stream);
  zero_f((float*)smax, (long long)N * 4, stream);
  zero_f(den, (long long)N * 4, stream);
  zero_f(acc, (long long)N * 20, stream);
  zero_f((float*)pooled, 128 * 20, stream);

  // ---- self-loop edge attr fill = mean(edge_attr) ----
  k_ea_mean<<<1024, 256, 0, stream>>>(ea, E, 1.0f / (float)E, mean3);

  // ---- layer 1: GATv2 (heads=3, ch=3), input 64 ----
  k_gemm_dual_wmma<64, 9><<<512, 256, 0, stream>>>(x, Wl1, Wr1, xl, xr, N);
  k_edge_score<3, 3><<<EB, 256, 0, stream>>>(src, dst, ea, mean3, xl, xr, We1,
                                             att1, sbuf, smax, E, Etot);
  k_edge_exp<3><<<EB, 256, 0, stream>>>(dst, smax, sbuf, den, E, Etot);
  k_edge_aggr<3, 3><<<EB, 256, 0, stream>>>(src, dst, xl, sbuf, den, acc, E, Etot);
  k_bias_relu<<<1024, 256, 0, stream>>>(acc, b1, h1, N, 9);

  // ---- reset accumulators for layer 2 ----
  zero_f((float*)smax, (long long)N * 4, stream);
  zero_f(den, (long long)N * 4, stream);
  zero_f(acc, (long long)N * 20, stream);

  // ---- layer 2: GATv2 (heads=4, ch=5), input 9 ----
  k_gemm_dual_wmma<9, 20><<<512, 256, 0, stream>>>(h1, Wl2, Wr2, xl, xr, N);
  k_edge_score<4, 5><<<EB, 256, 0, stream>>>(src, dst, ea, mean3, xl, xr, We2,
                                             att2, sbuf, smax, E, Etot);
  k_edge_exp<4><<<EB, 256, 0, stream>>>(dst, smax, sbuf, den, E, Etot);
  k_edge_aggr<4, 5><<<EB, 256, 0, stream>>>(src, dst, xl, sbuf, den, acc, E, Etot);

  // ---- relu + global max pool + MLP head ----
  k_bias_relu_pool<<<1024, 256, 0, stream>>>(acc, b2, batch, pooled, N);
  k_mlp<<<1, 128, 0, stream>>>(pooled, fw1, fb1, fw2, fb2, (float*)d_out, 128);
}
